// FormulaTransformer_13743895347625
// MI455X (gfx1250) — matrix operand order repaired and checked by reading, not compiled, not run on hardware
//
#include <hip/hip_runtime.h>
#include <hip/hip_bf16.h>

// ---------------------------------------------------------------------------
// FormulaTransformer for MI455X (gfx1250): all GEMMs via v_wmma_f32_16x16x32_f16
// B=4 N=256 F=17 H=256 NH=8 DH=32 L=4, INPUT_DIM=52 (padded to 64)
// ---------------------------------------------------------------------------

typedef __attribute__((ext_vector_type(16))) _Float16 v16h;
typedef __attribute__((ext_vector_type(8)))  _Float16 v8h;
typedef __attribute__((ext_vector_type(8)))  float    v8f;

union V16H { v16h v; v8h h2[2]; };

// ---- WMMA fragment helpers (wave32 layouts per CDNA5 ISA 7.12.2) ----------

// A-matrix 16x32 f16, row-major source, row stride lda (elements).
// lane L: row = L&15, khalf = L>>4; halfs[0..7]=A[r, 8*kh+0..7], halfs[8..15]=A[r,16+8*kh+0..7]
__device__ __forceinline__ v16h ldA(const _Float16* base, int lda) {
  int lane = threadIdx.x & 31;
  int r = lane & 15, kh = lane >> 4;
  const _Float16* p = base + (size_t)r * lda + kh * 8;
  V16H u;
  u.h2[0] = *(const v8h*)p;
  u.h2[1] = *(const v8h*)(p + 16);
  return u.v;
}

// B-matrix 32x16 f16 from a [N,K] (col-of-output-major) source: per lane,
// col n = L&15, khalf = L>>4 -> halfs[0..7]=K 8*kh+0..7, halfs[8..15]=K 16+8*kh+0..7.
__device__ __forceinline__ v16h ldB(const _Float16* base, int ldb) {
  int lane = threadIdx.x & 31;
  int n = lane & 15, kh = lane >> 4;
  const _Float16* p = base + (size_t)n * ldb + kh * 8;
  V16H u;
  u.h2[0] = *(const v8h*)p;
  u.h2[1] = *(const v8h*)(p + 16);
  return u.v;
}

__device__ __forceinline__ v8f wmma16(v16h a, v16h b, v8f c) {
  return __builtin_amdgcn_wmma_f32_16x16x32_f16(false, a, false, b, (short)0, c,
                                                false, false);
}

// ---- weight convert+transpose: W[K,N] f32 -> Wt[N,Kpad] f16 (zero padded) --
__global__ void wtrans_kernel(const float* __restrict__ W, _Float16* __restrict__ Wt,
                              int K, int N, int Kpad) {
  int n = blockIdx.x;
  for (int k = threadIdx.x; k < Kpad; k += blockDim.x)
    Wt[(size_t)n * Kpad + k] = (k < K) ? (_Float16)W[(size_t)k * N + n] : (_Float16)0.f;
}

// ---- feature build: x0 f16 [B*N, 64] -------------------------------------
__global__ void feat_kernel(const float* __restrict__ fv, const float* __restrict__ intens,
                            const int* __restrict__ types, const int* __restrict__ ion,
                            _Float16* __restrict__ X0) {
  __shared__ int cls;
  __shared__ float fc[17];
  int b = blockIdx.x, i = threadIdx.x;
  if (types[b * 256 + i] == 3) cls = i;
  __syncthreads();
  if (i < 17) fc[i] = fv[(size_t)(b * 256 + cls) * 17 + i];
  __syncthreads();
  const float* fr = fv + (size_t)(b * 256 + i) * 17;
  float tmp[64];
#pragma unroll
  for (int t = 0; t < 64; ++t) tmp[t] = 0.f;
  for (int f = 0; f < 17; ++f) { tmp[f] = fr[f]; tmp[17 + f] = fc[f] - fr[f]; }
  tmp[34 + types[b * 256 + i]] = 1.f;
  tmp[38 + ion[b * 256 + i]] = 1.f;
  tmp[46] = intens[b * 256 + i];
  _Float16* row = X0 + (size_t)(b * 256 + i) * 64;
  for (int t = 0; t < 64; ++t) row[t] = (_Float16)tmp[t];
}

// ---- |form_diffs| padded to K=32: Dabs f16 [B*N*N, 32] --------------------
__global__ void dabs_kernel(const float* __restrict__ fv, _Float16* __restrict__ D) {
  __shared__ float fi[17];
  int b = blockIdx.y, i = blockIdx.x, j = threadIdx.x;
  if (j < 17) fi[j] = fv[(size_t)(b * 256 + i) * 17 + j];
  __syncthreads();
  const float* fj = fv + (size_t)(b * 256 + j) * 17;
  _Float16* out = D + ((size_t)(b * 256 + i) * 256 + j) * 32;
  for (int f = 0; f < 17; ++f) out[f] = (_Float16)fabsf(fi[f] - fj[f]);
  for (int f = 17; f < 32; ++f) out[f] = (_Float16)0.f;
}

// ---- fused 2-layer MLP (Linear->ReLU->Linear->ReLU), hidden staged in LDS -
// A f16 [M,K1], W1t f16 [256,K1], W2t f16 [256,256]; 16 rows per block, 8 waves
__global__ __launch_bounds__(256) void mlp2_kernel(
    const _Float16* __restrict__ A, const _Float16* __restrict__ W1t,
    const float* __restrict__ b1, const _Float16* __restrict__ W2t,
    const float* __restrict__ b2, _Float16* __restrict__ Of16,
    float* __restrict__ Of32, int K1) {
  __shared__ _Float16 hsh[16 * 256];
  int tid = threadIdx.x, wave = tid >> 5, lane = tid & 31;
  int n = lane & 15, mh = lane >> 4;
  size_t m0 = (size_t)blockIdx.x * 16;
  const _Float16* Ab = A + m0 * K1;
  // stage 1: h = relu(A@W1 + b1) -> LDS
  for (int nt = wave * 2; nt < wave * 2 + 2; ++nt) {
    v8f c = {};
    for (int ks = 0; ks < K1 / 32; ++ks) {
      v16h a = ldA(Ab + ks * 32, K1);
      v16h bf = ldB(W1t + (size_t)(nt * 16) * K1 + ks * 32, K1);
      c = wmma16(a, bf, c);
    }
    float bv = b1[nt * 16 + n];
#pragma unroll
    for (int v = 0; v < 8; ++v) {
      float val = fmaxf(c[v] + bv, 0.f);
      hsh[(mh * 8 + v) * 256 + nt * 16 + n] = (_Float16)val;
    }
  }
  __syncthreads();
  // stage 2: out = relu(h@W2 + b2)
  for (int nt = wave * 2; nt < wave * 2 + 2; ++nt) {
    v8f c = {};
    for (int ks = 0; ks < 8; ++ks) {
      v16h a = ldA(&hsh[ks * 32], 256);
      v16h bf = ldB(W2t + (size_t)(nt * 16) * 256 + ks * 32, 256);
      c = wmma16(a, bf, c);
    }
    float bv = b2[nt * 16 + n];
#pragma unroll
    for (int v = 0; v < 8; ++v) {
      float val = fmaxf(c[v] + bv, 0.f);
      size_t row = m0 + mh * 8 + v;
      Of16[row * 256 + nt * 16 + n] = (_Float16)val;
      if (Of32) Of32[row * 256 + nt * 16 + n] = val;
    }
  }
}

// ---- generic GEMM: C = act(A[M,K] @ W[K,Nc] + bias), Wt pre-transposed ----
__global__ __launch_bounds__(256) void gemm_kernel(
    const _Float16* __restrict__ A, const _Float16* __restrict__ Wt,
    const float* __restrict__ bias, float* __restrict__ Cf32,
    _Float16* __restrict__ Cf16, int K, int Ncols, int relu) {
  int tid = threadIdx.x, wave = tid >> 5, lane = tid & 31;
  size_t m0 = (size_t)blockIdx.x * 16;
  int n0 = blockIdx.y * 256 + wave * 32;
  const _Float16* Ab = A + m0 * K;
  v8f c0 = {}, c1 = {};
  for (int ks = 0; ks < K / 32; ++ks) {
    __builtin_prefetch(Ab + (ks + 1) * 32, 0, 0);  // global_prefetch_b8
    v16h a = ldA(Ab + ks * 32, K);
    v16h bf0 = ldB(Wt + (size_t)n0 * K + ks * 32, K);
    c0 = wmma16(a, bf0, c0);
    v16h bf1 = ldB(Wt + (size_t)(n0 + 16) * K + ks * 32, K);
    c1 = wmma16(a, bf1, c1);
  }
  int n = lane & 15, mh = lane >> 4;
  for (int half = 0; half < 2; ++half) {
    v8f c = half ? c1 : c0;
    int nc = n0 + half * 16 + n;
    float bv = bias[nc];
#pragma unroll
    for (int v = 0; v < 8; ++v) {
      float val = c[v] + bv;
      if (relu) val = fmaxf(val, 0.f);
      size_t row = m0 + mh * 8 + v;
      if (Cf32) Cf32[row * Ncols + nc] = val;
      if (Cf16) Cf16[row * Ncols + nc] = (_Float16)val;
    }
  }
}

// ---- V transpose: vt[b][h][d][j] f16 --------------------------------------
__global__ void vtrans_kernel(const _Float16* __restrict__ qkvh, _Float16* __restrict__ vt) {
  int idx = blockIdx.x * 256 + threadIdx.x;  // ((b*8+h)*32+d)*256 + j
  int j = idx & 255, d = (idx >> 8) & 31, h = (idx >> 13) & 7, b = idx >> 16;
  vt[idx] = qkvh[(size_t)(b * 256 + j) * 768 + 512 + h * 32 + d];
}

// ---- attention: per (ichunk, head, batch); scores WMMA + pw bias + softmax
__global__ __launch_bounds__(256) void attn_kernel(
    const float* __restrict__ qkvf, const _Float16* __restrict__ qkvh,
    const _Float16* __restrict__ vt, const _Float16* __restrict__ pw,
    const int* __restrict__ num_peaks, _Float16* __restrict__ o) {
  __shared__ float sc[32 * 256];
  __shared__ _Float16 atn[32 * 256];
  int b = blockIdx.z, h = blockIdx.y, ic = blockIdx.x;
  int tid = threadIdx.x, wave = tid >> 5, lane = tid & 31;
  int i0 = ic * 32;
  // 1) q @ k^T -> sc (fp32)
  for (int t = wave; t < 32; t += 8) {
    int itile = t >> 4, jt = t & 15;
    const _Float16* aab = qkvh + (size_t)(b * 256 + i0 + itile * 16) * 768 + h * 32;
    v16h a = ldA(aab, 768);
    const _Float16* bb = qkvh + (size_t)(b * 256 + jt * 16) * 768 + 256 + h * 32;
    v16h bf = ldB(bb, 768);
    v8f c = {};
    c = wmma16(a, bf, c);
    int n = lane & 15, mh = lane >> 4;
#pragma unroll
    for (int v = 0; v < 8; ++v)
      sc[(itile * 16 + mh * 8 + v) * 256 + jt * 16 + n] = c[v];
  }
  __syncthreads();
  // 2) + pairwise bias, * scale, mask padding
  int np = num_peaks[b];
  {
    int il = tid >> 3, sub = tid & 7;
    int i = i0 + il;
    float qreg[32];
    const float* qp = qkvf + (size_t)(b * 256 + i) * 768 + h * 32;
#pragma unroll
    for (int d = 0; d < 32; ++d) qreg[d] = qp[d];
    const _Float16* pwrow = pw + ((size_t)(b * 256 + i) * 256) * 256 + h * 32;
    for (int jj = 0; jj < 32; ++jj) {
      int j = sub * 32 + jj;
      const _Float16* pp = pwrow + (size_t)j * 256;
      float acc = 0.f;
#pragma unroll
      for (int d = 0; d < 32; ++d) acc += qreg[d] * (float)pp[d];
      float s = (sc[il * 256 + j] + acc) * 0.17677669529663687f;  // 1/sqrt(32)
      if (j >= np) s = -1e9f;
      sc[il * 256 + j] = s;
    }
  }
  __syncthreads();
  // 3) softmax over j (one wave per 4 rows, wave32 shuffles)
  for (int rr = 0; rr < 4; ++rr) {
    int il = wave * 4 + rr;
    float vals[8];
    float mx = -3.4e38f;
#pragma unroll
    for (int t = 0; t < 8; ++t) {
      vals[t] = sc[il * 256 + lane + 32 * t];
      mx = fmaxf(mx, vals[t]);
    }
    for (int off = 16; off; off >>= 1) mx = fmaxf(mx, __shfl_xor(mx, off, 32));
    float sm = 0.f;
#pragma unroll
    for (int t = 0; t < 8; ++t) { vals[t] = expf(vals[t] - mx); sm += vals[t]; }
    for (int off = 16; off; off >>= 1) sm += __shfl_xor(sm, off, 32);
    float inv = 1.f / sm;
#pragma unroll
    for (int t = 0; t < 8; ++t)
      atn[il * 256 + lane + 32 * t] = (_Float16)(vals[t] * inv);
  }
  __syncthreads();
  // 4) attn @ v (K = 256, 8 WMMA steps); waves 0..3 each own a 16x16 tile
  if (wave < 4) {
    int itile = wave >> 1, dt = wave & 1;
    v8f c = {};
    for (int ks = 0; ks < 8; ++ks) {
      v16h a = ldA(&atn[(itile * 16) * 256 + ks * 32], 256);
      const _Float16* bb = vt + ((size_t)((b * 8 + h) * 32 + dt * 16)) * 256 + ks * 32;
      v16h bf = ldB(bb, 256);
      c = wmma16(a, bf, c);
    }
    int n = lane & 15, mh = lane >> 4;
#pragma unroll
    for (int v = 0; v < 8; ++v) {
      int i = i0 + itile * 16 + mh * 8 + v;
      o[(size_t)(b * 256 + i) * 256 + h * 32 + dt * 16 + n] = (_Float16)c[v];
    }
  }
}

// ---- residual + LayerNorm (one wave per 256-wide row) ---------------------
__global__ __launch_bounds__(256) void add_ln_kernel(
    const float* __restrict__ X, const float* __restrict__ T,
    const float* __restrict__ g, const float* __restrict__ bt,
    float* __restrict__ Xo, _Float16* __restrict__ Xh) {
  int wave = threadIdx.x >> 5, lane = threadIdx.x & 31;
  int row = blockIdx.x * 8 + wave;
  const float* xr = X + (size_t)row * 256;
  const float* tr = T + (size_t)row * 256;
  float v[8];
  float s = 0.f;
#pragma unroll
  for (int t = 0; t < 8; ++t) { v[t] = xr[lane + 32 * t] + tr[lane + 32 * t]; s += v[t]; }
  for (int off = 16; off; off >>= 1) s += __shfl_xor(s, off, 32);
  float mean = s * (1.f / 256.f);
  float q = 0.f;
#pragma unroll
  for (int t = 0; t < 8; ++t) { float d = v[t] - mean; q += d * d; }
  for (int off = 16; off; off >>= 1) q += __shfl_xor(q, off, 32);
  float inv = rsqrtf(q * (1.f / 256.f) + 1e-5f);
#pragma unroll
  for (int t = 0; t < 8; ++t) {
    int cc = lane + 32 * t;
    float y = (v[t] - mean) * inv * g[cc] + bt[cc];
    Xo[(size_t)row * 256 + cc] = y;
    Xh[(size_t)row * 256 + cc] = (_Float16)y;
  }
}

// ---- intensity pooling: out[b,d] = sum_{n<np} x*intens / (sum(intens)+eps) -
__global__ void pool_kernel(const float* __restrict__ X, const float* __restrict__ intens,
                            const int* __restrict__ num_peaks, float* __restrict__ out) {
  __shared__ float red[256];
  __shared__ float S;
  int b = blockIdx.x, t = threadIdx.x;
  red[t] = intens[b * 256 + t];
  __syncthreads();
  for (int s = 128; s; s >>= 1) {
    if (t < s) red[t] += red[t + s];
    __syncthreads();
  }
  if (t == 0) S = red[0];
  __syncthreads();
  int np = num_peaks[b];
  float acc = 0.f;
  for (int n = 0; n < np; ++n)
    acc += X[(size_t)(b * 256 + n) * 256 + t] * intens[b * 256 + n];
  out[b * 256 + t] = acc / (S + 1e-9f);
}

// ---------------------------------------------------------------------------
extern "C" void kernel_launch(void* const* d_in, const int* in_sizes, int n_in,
                              void* d_out, int out_size, void* d_ws, size_t ws_size,
                              hipStream_t stream) {
  (void)in_sizes; (void)n_in; (void)out_size; (void)ws_size;
  const int B = 4, N = 256, H = 256, NH = 8;
  const int BN = B * N;

  // inputs, setup_inputs() insertion order (nested dicts in insertion order)
  const float* form_vec  = (const float*)d_in[0];
  const float* intens    = (const float*)d_in[1];
  const int*   num_peaks = (const int*)d_in[2];
  const int*   types     = (const int*)d_in[3];
  /* instruments d_in[4] unused (embed_instrument=False) */
  const int*   ion_vec   = (const int*)d_in[5];
  const float* mlp_w1 = (const float*)d_in[6];
  const float* mlp_b1 = (const float*)d_in[7];
  const float* mlp_w2 = (const float*)d_in[8];
  const float* mlp_b2 = (const float*)d_in[9];
  const float* pw_w1  = (const float*)d_in[10];
  const float* pw_b1  = (const float*)d_in[11];
  const float* pw_w2  = (const float*)d_in[12];
  const float* pw_b2  = (const float*)d_in[13];
  const float* Lw[4][12];
  for (int l = 0; l < 4; ++l)
    for (int j = 0; j < 12; ++j) Lw[l][j] = (const float*)d_in[14 + 12 * l + j];
  // per-layer order: wqkv,bqkv,wo,bo,ffn_w1,ffn_b1,ffn_w2,ffn_b2,ln1_g,ln1_b,ln2_g,ln2_b

  // workspace carve-up (deterministic)
  char* wp = (char*)d_ws;
  auto alloc = [&](size_t bytes) -> void* {
    void* p = (void*)wp;
    wp += (bytes + 255) & ~(size_t)255;
    return p;
  };
  _Float16* X0f16   = (_Float16*)alloc((size_t)BN * 64 * 2);
  _Float16* Xf16    = (_Float16*)alloc((size_t)BN * 256 * 2);
  float*    Xf32    = (float*)alloc((size_t)BN * 256 * 4);
  float*    Tmp     = (float*)alloc((size_t)BN * 256 * 4);
  float*    qkv_f32 = (float*)alloc((size_t)BN * 768 * 4);
  _Float16* qkv_f16 = (_Float16*)alloc((size_t)BN * 768 * 2);
  _Float16* o_f16   = (_Float16*)alloc((size_t)BN * 256 * 2);
  _Float16* Hff16   = (_Float16*)alloc((size_t)BN * 1024 * 2);
  _Float16* vt      = (_Float16*)alloc((size_t)B * NH * 32 * 256 * 2);
  _Float16* mlp_w1t = (_Float16*)alloc((size_t)256 * 64 * 2);
  _Float16* mlp_w2t = (_Float16*)alloc((size_t)256 * 256 * 2);
  _Float16* pw_w1t  = (_Float16*)alloc((size_t)256 * 32 * 2);
  _Float16* pw_w2t  = (_Float16*)alloc((size_t)256 * 256 * 2);
  _Float16* wqkvt[4]; _Float16* wot[4]; _Float16* w1t[4]; _Float16* w2t[4];
  for (int l = 0; l < 4; ++l) {
    wqkvt[l] = (_Float16*)alloc((size_t)768 * 256 * 2);
    wot[l]   = (_Float16*)alloc((size_t)256 * 256 * 2);
    w1t[l]   = (_Float16*)alloc((size_t)1024 * 256 * 2);
    w2t[l]   = (_Float16*)alloc((size_t)256 * 1024 * 2);
  }
  _Float16* Dabs = (_Float16*)alloc((size_t)B * N * N * 32 * 2);   // 16 MB
  _Float16* PW   = (_Float16*)alloc((size_t)B * N * N * 256 * 2);  // 134 MB

  // 1) weight transpose+f16 convert
  wtrans_kernel<<<256, 256, 0, stream>>>(mlp_w1, mlp_w1t, 52, 256, 64);
  wtrans_kernel<<<256, 256, 0, stream>>>(mlp_w2, mlp_w2t, 256, 256, 256);
  wtrans_kernel<<<256, 256, 0, stream>>>(pw_w1, pw_w1t, 17, 256, 32);
  wtrans_kernel<<<256, 256, 0, stream>>>(pw_w2, pw_w2t, 256, 256, 256);
  for (int l = 0; l < 4; ++l) {
    wtrans_kernel<<<768, 256, 0, stream>>>(Lw[l][0], wqkvt[l], 256, 768, 256);
    wtrans_kernel<<<256, 256, 0, stream>>>(Lw[l][2], wot[l], 256, 256, 256);
    wtrans_kernel<<<1024, 256, 0, stream>>>(Lw[l][4], w1t[l], 256, 1024, 256);
    wtrans_kernel<<<256, 256, 0, stream>>>(Lw[l][6], w2t[l], 1024, 256, 1024);
  }

  // 2) peak features + peak MLP -> x
  feat_kernel<<<B, 256, 0, stream>>>(form_vec, intens, types, ion_vec, X0f16);
  mlp2_kernel<<<BN / 16, 256, 0, stream>>>(X0f16, mlp_w1t, mlp_b1, mlp_w2t, mlp_b2,
                                           Xf16, Xf32, 64);

  // 3) pairwise features + fused pairwise MLP -> PW (f16, [B*N*N, 256])
  dabs_kernel<<<dim3(N, B), 256, 0, stream>>>(form_vec, Dabs);
  mlp2_kernel<<<(B * N * N) / 16, 256, 0, stream>>>(Dabs, pw_w1t, pw_b1, pw_w2t,
                                                    pw_b2, PW, (float*)nullptr, 32);

  // 4) transformer layers
  for (int l = 0; l < 4; ++l) {
    gemm_kernel<<<dim3(BN / 16, 3), 256, 0, stream>>>(Xf16, wqkvt[l], Lw[l][1],
                                                      qkv_f32, qkv_f16, 256, 768, 0);
    vtrans_kernel<<<(B * NH * 32 * 256) / 256, 256, 0, stream>>>(qkv_f16, vt);
    attn_kernel<<<dim3(8, NH, B), 256, 0, stream>>>(qkv_f32, qkv_f16, vt, PW,
                                                    num_peaks, o_f16);
    gemm_kernel<<<dim3(BN / 16, 1), 256, 0, stream>>>(o_f16, wot[l], Lw[l][3],
                                                      Tmp, (_Float16*)nullptr, 256, 256, 0);
    add_ln_kernel<<<BN / 8, 256, 0, stream>>>(Xf32, Tmp, Lw[l][8], Lw[l][9], Xf32, Xf16);
    gemm_kernel<<<dim3(BN / 16, 4), 256, 0, stream>>>(Xf16, w1t[l], Lw[l][5],
                                                      (float*)nullptr, Hff16, 256, 1024, 1);
    gemm_kernel<<<dim3(BN / 16, 1), 256, 0, stream>>>(Hff16, w2t[l], Lw[l][7],
                                                      Tmp, (_Float16*)nullptr, 1024, 256, 0);
    add_ln_kernel<<<BN / 8, 256, 0, stream>>>(Xf32, Tmp, Lw[l][10], Lw[l][11], Xf32, Xf16);
  }

  // 5) intensity pooling -> out [B, H]
  pool_kernel<<<B, 256, 0, stream>>>(Xf32, intens, num_peaks, (float*)d_out);
}
